// Dynamic_conv2d_const_channel_26276609917085
// MI455X (gfx1250) — matrix operand order repaired and checked by reading, not compiled, not run on hardware
//
#include <hip/hip_runtime.h>
#include <hip/hip_bf16.h>
#include <math.h>

typedef __attribute__((ext_vector_type(16))) __bf16 v16bf;
typedef __attribute__((ext_vector_type(8)))  float  v8f;

#define B_   32
#define C_   256
#define H_   56
#define W_   56
#define K_   4
#define HID  65
#define HW_  (H_*W_)
// K-dim of implicit GEMM: 8 i_outer blocks * 9 taps * 32 channels = 2304
#define KSTEPS 72
// fragment buffer: B * 16 o-tiles * 72 ksteps * 32 lanes * 16 bf16
#define FRAG_ELEMS (32u*16u*72u*32u*16u)   // 18,874,368 ushorts = 37.75 MB

__device__ __forceinline__ unsigned short f2bf(float f) {
    unsigned int u = __float_as_uint(f);
    u += 0x7FFFu + ((u >> 16) & 1u);          // round-to-nearest-even
    return (unsigned short)(u >> 16);
}

// ---------------- Kernel 1: global average pool -> pooled[B*C] ----------------
__global__ __launch_bounds__(256) void pool_kernel(const float* __restrict__ x,
                                                   float* __restrict__ pooled) {
    const int bc = blockIdx.x;                 // 0 .. B*C-1
    const float* p = x + (size_t)bc * HW_;
    float s = 0.f;
    for (int e = threadIdx.x; e < HW_; e += 256) s += p[e];
    __shared__ float red[256];
    red[threadIdx.x] = s;
    __syncthreads();
    for (int off = 128; off > 0; off >>= 1) {
        if (threadIdx.x < off) red[threadIdx.x] += red[threadIdx.x + off];
        __syncthreads();
    }
    if (threadIdx.x == 0) pooled[bc] = red[0] * (1.0f / (float)HW_);
}

// ------------- Kernel 2: attention MLP + softmax + aggregated bias -------------
__global__ __launch_bounds__(256) void attn_kernel(const float* __restrict__ pooled,
                                                   const float* __restrict__ fc1,
                                                   const float* __restrict__ fc2,
                                                   const float* __restrict__ fc2b,
                                                   const float* __restrict__ bias,
                                                   float* __restrict__ att,
                                                   float* __restrict__ aggb) {
    const int b = blockIdx.x;
    const int t = threadIdx.x;
    __shared__ float sp[C_];
    __shared__ float sh[HID];
    __shared__ float sa[K_];
    sp[t] = pooled[b * C_ + t];
    __syncthreads();
    if (t < HID) {
        float acc = 0.f;
        for (int c = 0; c < C_; ++c) acc += sp[c] * fc1[t * C_ + c];
        sh[t] = fmaxf(acc, 0.f);
    }
    __syncthreads();
    if (t < K_) {
        float acc = fc2b[t];
        for (int j = 0; j < HID; ++j) acc += sh[j] * fc2[t * HID + j];
        sa[t] = acc * (1.0f / 34.0f);          // logits / TEMPERATURE
    }
    __syncthreads();
    if (t == 0) {
        float m = fmaxf(fmaxf(sa[0], sa[1]), fmaxf(sa[2], sa[3]));
        float e0 = expf(sa[0] - m), e1 = expf(sa[1] - m);
        float e2 = expf(sa[2] - m), e3 = expf(sa[3] - m);
        float inv = 1.0f / (e0 + e1 + e2 + e3);
        sa[0] = e0 * inv; sa[1] = e1 * inv; sa[2] = e2 * inv; sa[3] = e3 * inv;
    }
    __syncthreads();
    if (t < K_) att[b * K_ + t] = sa[t];
    // aggregated per-sample bias: aggb[b, o] = sum_k att[b,k] * bias[k,o]
    float ab = 0.f;
    for (int k = 0; k < K_; ++k) ab += sa[k] * bias[k * C_ + t];
    aggb[b * C_ + t] = ab;
}

// ---- Kernel 3: aggregate weights -> bf16, pre-swizzled WMMA A fragments ----
// Layout: fragA[((b*16 + o16)*72 + kstep)*32*16 + lane*16 + j]
// A-matrix 16x32 bf16 lane layout:
//   lane<16:  row M=lane,    elem j -> K = j                        in {0-15}
//   lane>=16: row M=lane-16, elem j -> K = 16 + j                   in {16-31}
// kstep = i_outer*9 + khkw ; channel i = i_outer*32 + K ; tap = khkw
__global__ __launch_bounds__(256) void frag_kernel(const float* __restrict__ weight,
                                                   const float* __restrict__ att,
                                                   unsigned short* __restrict__ fragA) {
    const unsigned int gi = blockIdx.x * 256u + threadIdx.x;   // < FRAG_ELEMS
    const int j    = gi & 15;
    const int lane = (gi >> 4) & 31;
    unsigned int r = gi >> 9;
    const int kstep = r % KSTEPS; r /= KSTEPS;
    const int o16   = r & 15;
    const int b     = r >> 4;

    const int o  = o16 * 16 + (lane & 15);
    const int Kk = ((lane >> 4) ? 16 : 0) + j;
    const int io = kstep / 9, khkw = kstep % 9;
    const int i  = io * 32 + Kk;

    const float* a = att + b * K_;
    const size_t widx = ((size_t)o * C_ + i) * 9 + khkw;       // weight[k][o][i][kh][kw]
    const size_t kpitch = (size_t)C_ * C_ * 9;
    float v = a[0] * weight[widx]
            + a[1] * weight[widx + kpitch]
            + a[2] * weight[widx + 2 * kpitch]
            + a[3] * weight[widx + 3 * kpitch];
    fragA[gi] = f2bf(v);
}

// ---------------- Kernel 4: implicit-GEMM conv via WMMA bf16 ----------------
// Block: 256 threads = 8 waves. Block tile: 256 out-channels x 64 spatial (8x8)
// -> each (sample, spatial tile) of x is staged exactly once (halves x traffic
//    vs the 128-O version; x is the dominant HBM term at 23.3 TB/s).
// Wave (wm 0..3, wn 0..1) computes 64 O x 32 N as 4x2 tiles of 16x16.
// K loop: 8 channel-blocks x 9 taps, K=32 per wmma -> 8 wmma per kstep/wave.
__global__ __launch_bounds__(256) void conv_kernel(const float* __restrict__ x,
                                                   const unsigned short* __restrict__ fragA,
                                                   const float* __restrict__ aggb,
                                                   float* __restrict__ out) {
    const int b    = blockIdx.y;
    const int tile = blockIdx.x;            // 0..48 (7x7 tiles of 8x8)
    const int y0 = (tile / 7) * 8;
    const int x0 = (tile % 7) * 8;

    const int tid  = threadIdx.x;
    const int wave = tid >> 5;
    const int lane = tid & 31;
    const int wm = wave >> 1;               // 0..3 -> 64 O each
    const int wn = wave & 1;                // 0..1 -> 32 N each
    const int col    = lane & 15;
    const int ksplit = lane >> 4;

    // halo tile: 10x10 positions, 32 channels, channel-last bf16 (6400 B)
    __shared__ __align__(32) unsigned short lds_x[100 * 32];

    v8f acc[4][2];
    #pragma unroll
    for (int m = 0; m < 4; ++m)
        #pragma unroll
        for (int n = 0; n < 2; ++n)
            #pragma unroll
            for (int q = 0; q < 8; ++q) acc[m][n][q] = 0.f;

    for (int io = 0; io < 8; ++io) {
        // stage 32 channels of the 10x10 halo into LDS (coalesced over positions)
        for (int e = tid; e < 3200; e += 256) {
            const int c   = e / 100;
            const int pos = e - c * 100;
            const int py = pos / 10, px = pos - py * 10;
            const int yy = y0 + py - 1, xx = x0 + px - 1;
            float v = 0.f;
            if ((unsigned)yy < (unsigned)H_ && (unsigned)xx < (unsigned)W_)
                v = x[(((size_t)b * C_ + io * 32 + c) * H_ + yy) * W_ + xx];
            lds_x[pos * 32 + c] = f2bf(v);
        }
        __syncthreads();

        #pragma unroll
        for (int khkw = 0; khkw < 9; ++khkw) {
            const int dh = khkw / 3, dw = khkw - dh * 3;
            const int kstep = io * 9 + khkw;

            v16bf afrag[4], bfrag[2];
            #pragma unroll
            for (int m = 0; m < 4; ++m) {
                const size_t off =
                    ((((size_t)b * 16 + wm * 4 + m) * KSTEPS + kstep) * 32 + lane) * 16;
                afrag[m] = *reinterpret_cast<const v16bf*>(fragA + off);
            }
            #pragma unroll
            for (int n = 0; n < 2; ++n) {
                const int p  = wn * 32 + n * 16 + col;
                const int pr = p >> 3, pc = p & 7;
                const int off = ((pr + dh) * 10 + (pc + dw)) * 32 + ksplit * 16;
                bfrag[n] = *reinterpret_cast<const v16bf*>(lds_x + off);
            }
            #pragma unroll
            for (int m = 0; m < 4; ++m)
                #pragma unroll
                for (int n = 0; n < 2; ++n)
                    acc[m][n] = __builtin_amdgcn_wmma_f32_16x16x32_bf16(
                        false, afrag[m], false, bfrag[n],
                        (short)0, acc[m][n], false, false);
        }
        __syncthreads();
    }

    // epilogue: C layout (VGPR q: lanes 0-15 -> M=q, lanes 16-31 -> M=q+8)
    #pragma unroll
    for (int m = 0; m < 4; ++m) {
        #pragma unroll
        for (int n = 0; n < 2; ++n) {
            const int p  = wn * 32 + n * 16 + col;
            const int pr = p >> 3, pc = p & 7;
            const int yy = y0 + pr, xx = x0 + pc;
            #pragma unroll
            for (int q = 0; q < 8; ++q) {
                const int o = wm * 64 + m * 16 + ksplit * 8 + q;
                out[(((size_t)b * C_ + o) * H_ + yy) * W_ + xx] =
                    acc[m][n][q] + aggb[b * C_ + o];
            }
        }
    }
}

extern "C" void kernel_launch(void* const* d_in, const int* in_sizes, int n_in,
                              void* d_out, int out_size, void* d_ws, size_t ws_size,
                              hipStream_t stream) {
    const float* x    = (const float*)d_in[0];   // [32,256,56,56]
    const float* fc1  = (const float*)d_in[1];   // [65,256]
    const float* fc2  = (const float*)d_in[2];   // [4,65]
    const float* fc2b = (const float*)d_in[3];   // [4]
    const float* wgt  = (const float*)d_in[4];   // [4,256,256,3,3]
    const float* bias = (const float*)d_in[5];   // [4,256]
    float* out = (float*)d_out;                  // [32,256,56,56]

    char* ws = (char*)d_ws;
    unsigned short* fragA = (unsigned short*)ws;                     // 37.75 MB
    float* pooled = (float*)(ws + (size_t)FRAG_ELEMS * 2);           // 32 KB
    float* att    = pooled + B_ * C_;                                // 512 B
    float* aggb   = att + B_ * K_;                                   // 32 KB

    pool_kernel<<<B_ * C_, 256, 0, stream>>>(x, pooled);
    attn_kernel<<<B_, 256, 0, stream>>>(pooled, fc1, fc2, fc2b, bias, att, aggb);
    frag_kernel<<<FRAG_ELEMS / 256, 256, 0, stream>>>(wgt, att, fragA);
    conv_kernel<<<dim3(49, B_), 256, 0, stream>>>(x, fragA, aggb, out);
}
